// GAT_Surrogate_88562225643706
// MI455X (gfx1250) — compile-verified, matrix operand order checked by reading, not run on hardware
//
#include <hip/hip_runtime.h>
#include <hip/hip_bf16.h>

// ---------------------------------------------------------------------------
// Types for WMMA (gfx1250, wave32)
// ---------------------------------------------------------------------------
typedef __attribute__((ext_vector_type(16))) __bf16 v16bf;
typedef __attribute__((ext_vector_type(8)))  __bf16 bf16x8;
typedef __attribute__((ext_vector_type(8)))  float  v8f;

#define NEG_SLOPE 0.2f

// ---------------------------------------------------------------------------
// Helpers
// ---------------------------------------------------------------------------
__device__ __forceinline__ unsigned short f2bf_rne(float f) {
    unsigned u = __float_as_uint(f);
    unsigned r = u + 0x7fffu + ((u >> 16) & 1u);
    return (unsigned short)(r >> 16);
}

// float atomic max via integer atomics (works for all sign combinations,
// init value must be -inf)
__device__ __forceinline__ void atomicMaxFloat(float* addr, float val) {
    if (val >= 0.0f) {
        atomicMax((int*)addr, __float_as_int(val));
    } else {
        atomicMin((unsigned int*)addr, __float_as_uint(val));
    }
}

// Load one 16-element bf16 WMMA fragment (two contiguous 16B chunks at
// +0 and +16 elements, per the ISA 16-bit fragment layout)
__device__ __forceinline__ v16bf load_frag(const __bf16* p) {
    bf16x8 lo = *(const bf16x8*)(p);
    bf16x8 hi = *(const bf16x8*)(p + 16);
    return __builtin_shufflevector(lo, hi,
        0,1,2,3,4,5,6,7,8,9,10,11,12,13,14,15);
}

// ---------------------------------------------------------------------------
// Generic fill
// ---------------------------------------------------------------------------
__global__ void fill_f32_kernel(float* __restrict__ p, float v, long long n) {
    long long i = (long long)blockIdx.x * blockDim.x + threadIdx.x;
    if (i < n) p[i] = v;
}

// ---------------------------------------------------------------------------
// f32 -> bf16 elementwise convert
// ---------------------------------------------------------------------------
__global__ void f32_to_bf16_kernel(const float* __restrict__ in,
                                   unsigned short* __restrict__ out,
                                   long long n) {
    long long i = (long long)blockIdx.x * blockDim.x + threadIdx.x;
    if (i < n) out[i] = f2bf_rne(in[i]);
}

// ---------------------------------------------------------------------------
// W [K, Nout] f32 (row-major)  ->  Wt [Nout, K] bf16 (row-major)
// ---------------------------------------------------------------------------
__global__ void transpose_to_bf16_kernel(const float* __restrict__ W,
                                         unsigned short* __restrict__ Wt,
                                         int K, int Nout) {
    long long i = (long long)blockIdx.x * blockDim.x + threadIdx.x;
    if (i >= (long long)K * Nout) return;
    int k = (int)(i / Nout);
    int n = (int)(i % Nout);
    Wt[(size_t)n * K + k] = f2bf_rne(W[i]);
}

// ---------------------------------------------------------------------------
// Encoder: out[N,128] = x[N,16] @ W[16,128] + b  (tiny K -> VALU)
// ---------------------------------------------------------------------------
__global__ void encoder_kernel(const float* __restrict__ x,
                               const float* __restrict__ W,
                               const float* __restrict__ b,
                               float* __restrict__ out, int N) {
    long long i = (long long)blockIdx.x * blockDim.x + threadIdx.x;
    if (i >= (long long)N * 128) return;
    int n = (int)(i >> 7);
    int c = (int)(i & 127);
    float acc = b[c];
#pragma unroll
    for (int k = 0; k < 16; ++k) acc += x[n * 16 + k] * W[k * 128 + c];
    out[i] = acc;
}

// ---------------------------------------------------------------------------
// WMMA bf16 GEMM:  C[M,Nout] = A[M,K] @ Bt[Nout,K]^T   (f32 accumulate)
//
// block = 128 threads = 4 waves; each wave computes a 32x32 output block
// (2 M-tiles x 2 N-tiles = 4 WMMAs per K-step, with 2 A-frag + 2 B-frag
// loads -> 2x the arithmetic intensity of a 1-tile wave).
// Block covers 32 rows x 128 cols.  M%32==0, K%32==0, Nout%128==0.
//
// A-fragment per-lane layout (ISA 7.12.2, 16-bit A 16x32):
//   lane L: row = L&15, kh = L>>4
//   v16bf elements [0..7]  = K = k0 + kh*8 + {0..7}
//   v16bf elements [8..15] = K = k0 + 16 + kh*8 + {0..7}
// B loaded from the pre-transposed weight with the mirrored layout
// (col = L&15, same K packing).  D layout: acc[r] -> row m_tile + r + 8*kh.
// ---------------------------------------------------------------------------
__global__ void wmma_gemm_bf16_kernel(const __bf16* __restrict__ A,
                                      const __bf16* __restrict__ Bt,
                                      float* __restrict__ C,
                                      int K, int Nout) {
    const int lane = threadIdx.x & 31;
    const int wv   = threadIdx.x >> 5;          // 0..3
    const int n0   = blockIdx.x * 128 + wv * 32;
    const int m0   = blockIdx.y * 32;
    const int r    = lane & 15;
    const int kh   = lane >> 4;

    const __bf16* arow0 = A  + (size_t)(m0 + r) * K + kh * 8;
    const __bf16* arow1 = arow0 + (size_t)16 * K;
    const __bf16* brow0 = Bt + (size_t)(n0 + r) * K + kh * 8;
    const __bf16* brow1 = brow0 + (size_t)16 * K;

    v8f acc00 = {}, acc01 = {}, acc10 = {}, acc11 = {};
    for (int k0 = 0; k0 < K; k0 += 32) {
        v16bf a0 = load_frag(arow0 + k0);
        v16bf a1 = load_frag(arow1 + k0);
        v16bf b0 = load_frag(brow0 + k0);
        v16bf b1 = load_frag(brow1 + k0);
        acc00 = __builtin_amdgcn_wmma_f32_16x16x32_bf16(
                    false, a0, false, b0, (short)0, acc00, false, false);
        acc01 = __builtin_amdgcn_wmma_f32_16x16x32_bf16(
                    false, a0, false, b1, (short)0, acc01, false, false);
        acc10 = __builtin_amdgcn_wmma_f32_16x16x32_bf16(
                    false, a1, false, b0, (short)0, acc10, false, false);
        acc11 = __builtin_amdgcn_wmma_f32_16x16x32_bf16(
                    false, a1, false, b1, (short)0, acc11, false, false);
    }

    float* c00 = C + (size_t)(m0 + kh * 8) * Nout + n0 + r;
    float* c10 = c00 + (size_t)16 * Nout;
#pragma unroll
    for (int rr = 0; rr < 8; ++rr) {
        c00[(size_t)rr * Nout]      = acc00[rr];
        c00[(size_t)rr * Nout + 16] = acc01[rr];
        c10[(size_t)rr * Nout]      = acc10[rr];
        c10[(size_t)rr * Nout + 16] = acc11[rr];
    }
}

// ---------------------------------------------------------------------------
// Attention coefficients: asrc[n,h] = sum_c hlin[n,h,c]*a_s[h,c], same adst
// ---------------------------------------------------------------------------
__global__ void alpha_kernel(const float* __restrict__ hlin,
                             const float* __restrict__ a_s,
                             const float* __restrict__ a_d,
                             float* __restrict__ asrc,
                             float* __restrict__ adst,
                             int N, int H, int C) {
    long long i = (long long)blockIdx.x * blockDim.x + threadIdx.x;
    if (i >= (long long)N * H) return;
    int n = (int)(i / H);
    int h = (int)(i % H);
    const float* hp  = hlin + ((size_t)n * H + h) * C;
    const float* asp = a_s + (size_t)h * C;
    const float* adp = a_d + (size_t)h * C;
    float s = 0.f, d = 0.f;
    for (int c = 0; c < C; ++c) {
        float v = hp[c];
        s += v * asp[c];
        d += v * adp[c];
    }
    asrc[i] = s;
    adst[i] = d;
}

// ---------------------------------------------------------------------------
// Edge pass 1: e = leaky_relu(asrc[src]+adst[dst]); store e; segment max
// ---------------------------------------------------------------------------
__global__ void edge_max_kernel(const int* __restrict__ ei, int Eorig, int N,
                                const float* __restrict__ asrc,
                                const float* __restrict__ adst,
                                float* __restrict__ eval,
                                float* __restrict__ mmax, int H) {
    int e = blockIdx.x * blockDim.x + threadIdx.x;
    int E = Eorig + N;
    if (e >= E) return;
    int s, d;
    if (e < Eorig) { s = ei[e]; d = ei[Eorig + e]; }
    else           { s = d = e - Eorig; }
    for (int h = 0; h < H; ++h) {
        float v = asrc[(size_t)s * H + h] + adst[(size_t)d * H + h];
        v = v > 0.f ? v : NEG_SLOPE * v;
        eval[(size_t)e * H + h] = v;
        atomicMaxFloat(&mmax[(size_t)d * H + h], v);
    }
}

// ---------------------------------------------------------------------------
// Edge pass 2: p = exp(e - max[dst]); store p; denom[dst] += p
// ---------------------------------------------------------------------------
__global__ void edge_exp_kernel(const int* __restrict__ ei, int Eorig, int N,
                                float* __restrict__ eval,
                                const float* __restrict__ mmax,
                                float* __restrict__ denom, int H) {
    int e = blockIdx.x * blockDim.x + threadIdx.x;
    int E = Eorig + N;
    if (e >= E) return;
    int d = (e < Eorig) ? ei[Eorig + e] : (e - Eorig);
    for (int h = 0; h < H; ++h) {
        float p = __expf(eval[(size_t)e * H + h] - mmax[(size_t)d * H + h]);
        eval[(size_t)e * H + h] = p;
        atomicAdd(&denom[(size_t)d * H + h], p);
    }
}

// ---------------------------------------------------------------------------
// Message scatter: accum[dst,h,:] += hlin[src,h,:] * p/(denom[dst,h]+eps)
// one thread per (edge, 4-channel chunk)
// ---------------------------------------------------------------------------
__global__ void scatter_kernel(const int* __restrict__ ei, int Eorig, int N,
                               const float* __restrict__ hlin,
                               const float* __restrict__ eval,
                               const float* __restrict__ denom,
                               float* __restrict__ accum, int H, int C) {
    int units = (H * C) >> 2;  // float4 chunks per edge
    long long i = (long long)blockIdx.x * blockDim.x + threadIdx.x;
    long long total = (long long)(Eorig + N) * units;
    if (i >= total) return;
    int e  = (int)(i / units);
    int u  = (int)(i % units);
    int c4 = u * 4;
    int h  = c4 / C;
    int s, d;
    if (e < Eorig) { s = ei[e]; d = ei[Eorig + e]; }
    else           { s = d = e - Eorig; }
    float alpha = eval[(size_t)e * H + h] /
                  (denom[(size_t)d * H + h] + 1e-16f);
    const float4 v = *(const float4*)(hlin + (size_t)s * H * C + c4);
    float* o = accum + (size_t)d * H * C + c4;
    atomicAdd(o + 0, v.x * alpha);
    atomicAdd(o + 1, v.y * alpha);
    atomicAdd(o + 2, v.z * alpha);
    atomicAdd(o + 3, v.w * alpha);
}

// ---------------------------------------------------------------------------
// Epilogue: out = (accum + bias) [relu]; optionally emit bf16 copy
// ---------------------------------------------------------------------------
__global__ void post_kernel(float* __restrict__ accum,
                            const float* __restrict__ bias,
                            unsigned short* __restrict__ out_bf,
                            int N, int HC, int do_relu) {
    long long i = (long long)blockIdx.x * blockDim.x + threadIdx.x;
    if (i >= (long long)N * HC) return;
    int c = (int)(i % HC);
    float v = accum[i] + bias[c];
    if (do_relu) v = v > 0.f ? v : 0.f;
    accum[i] = v;
    if (out_bf) out_bf[i] = f2bf_rne(v);
}

// ---------------------------------------------------------------------------
// Decoder: out[N,3] = h[N,128] @ W[128,3] + b
// ---------------------------------------------------------------------------
__global__ void decoder_kernel(const float* __restrict__ h,
                               const float* __restrict__ W,
                               const float* __restrict__ b,
                               float* __restrict__ out, int N) {
    long long i = (long long)blockIdx.x * blockDim.x + threadIdx.x;
    if (i >= (long long)N * 3) return;
    int n = (int)(i / 3);
    int j = (int)(i % 3);
    float acc = b[j];
#pragma unroll 8
    for (int k = 0; k < 128; ++k) acc += h[n * 128 + k] * W[k * 3 + j];
    out[i] = acc;
}

// ---------------------------------------------------------------------------
// Host-side launch
// ---------------------------------------------------------------------------
static inline int nblk(long long n, int b) { return (int)((n + b - 1) / b); }

extern "C" void kernel_launch(void* const* d_in, const int* in_sizes, int n_in,
                              void* d_out, int out_size, void* d_ws, size_t ws_size,
                              hipStream_t stream) {
    const int N  = 20000;
    const int E0 = 320000;
    const int E  = E0 + N;

    const float* x     = (const float*)d_in[0];
    const int*   ei    = (const int*)  d_in[1];
    const float* enc_W = (const float*)d_in[2];
    const float* enc_b = (const float*)d_in[3];
    const float* g_W [3] = {(const float*)d_in[4],  (const float*)d_in[8],  (const float*)d_in[12]};
    const float* g_as[3] = {(const float*)d_in[5],  (const float*)d_in[9],  (const float*)d_in[13]};
    const float* g_ad[3] = {(const float*)d_in[6],  (const float*)d_in[10], (const float*)d_in[14]};
    const float* g_b [3] = {(const float*)d_in[7],  (const float*)d_in[11], (const float*)d_in[15]};
    const float* dec_W = (const float*)d_in[16];
    const float* dec_b = (const float*)d_in[17];
    float* out = (float*)d_out;

    // ---- workspace carve-up ----
    char* ws = (char*)d_ws;
    size_t off = 0;
    auto take = [&](size_t bytes) -> void* {
        void* p = ws + off;
        off += (bytes + 255) & ~(size_t)255;
        return p;
    };
    float*          buf0  = (float*)take((size_t)N * 512 * sizeof(float)); // input / accum
    float*          buf1  = (float*)take((size_t)N * 512 * sizeof(float)); // hlin
    unsigned short* bufbf = (unsigned short*)take((size_t)N * 512 * 2);    // bf16 input
    unsigned short* wt    = (unsigned short*)take((size_t)512 * 512 * 2);  // bf16 W^T
    float*          asrc  = (float*)take((size_t)N * 4 * sizeof(float));
    float*          adst  = (float*)take((size_t)N * 4 * sizeof(float));
    float*          mmax  = (float*)take((size_t)N * 4 * sizeof(float));
    float*          denom = (float*)take((size_t)N * 4 * sizeof(float));
    float*          eval  = (float*)take((size_t)E * 4 * sizeof(float));
    (void)ws_size; (void)in_sizes; (void)n_in; (void)out_size;

    const int B = 256;

    // ---- encoder: buf0 = x @ enc_W + enc_b  [N,128] ----
    encoder_kernel<<<nblk((long long)N * 128, B), B, 0, stream>>>(x, enc_W, enc_b, buf0, N);

    // ---- three GAT layers ----
    const int   K_in [3] = {128, 512, 512};
    const int   H_l  [3] = {4, 4, 1};
    const int   C_l  [3] = {128, 128, 128};
    const int   relu [3] = {1, 1, 0};

    for (int l = 0; l < 3; ++l) {
        const int K   = K_in[l];
        const int H   = H_l[l];
        const int C   = C_l[l];
        const int HC  = H * C;

        // bf16 copy of input, transposed bf16 weight
        f32_to_bf16_kernel<<<nblk((long long)N * K, B), B, 0, stream>>>(
            buf0, bufbf, (long long)N * K);
        transpose_to_bf16_kernel<<<nblk((long long)K * HC, B), B, 0, stream>>>(
            g_W[l], wt, K, HC);

        // WMMA GEMM: buf1[N,HC] = bufbf[N,K] @ wt[HC,K]^T
        // block = 128 threads (4 waves), wave tile = 32x32,
        // block tile = 32 rows x 128 cols
        dim3 ggrid(HC / 128, N / 32);
        wmma_gemm_bf16_kernel<<<ggrid, 128, 0, stream>>>(
            (const __bf16*)bufbf, (const __bf16*)wt, buf1, K, HC);

        // attention coefficients
        alpha_kernel<<<nblk((long long)N * H, B), B, 0, stream>>>(
            buf1, g_as[l], g_ad[l], asrc, adst, N, H, C);

        // init segment max / denom / output accumulator (buf0 is now free)
        fill_f32_kernel<<<nblk((long long)N * H, B), B, 0, stream>>>(
            mmax, -__builtin_inff(), (long long)N * H);
        fill_f32_kernel<<<nblk((long long)N * H, B), B, 0, stream>>>(
            denom, 0.f, (long long)N * H);
        fill_f32_kernel<<<nblk((long long)N * HC, B), B, 0, stream>>>(
            buf0, 0.f, (long long)N * HC);

        // segment softmax over incoming edges
        edge_max_kernel<<<nblk(E, B), B, 0, stream>>>(ei, E0, N, asrc, adst, eval, mmax, H);
        edge_exp_kernel<<<nblk(E, B), B, 0, stream>>>(ei, E0, N, eval, mmax, denom, H);

        // weighted message scatter
        long long sc_total = (long long)E * (HC >> 2);
        scatter_kernel<<<nblk(sc_total, B), B, 0, stream>>>(
            ei, E0, N, buf1, eval, denom, buf0, H, C);

        // bias + (relu) + bf16 copy for next layer (layer 3 needs no bf16)
        post_kernel<<<nblk((long long)N * HC, B), B, 0, stream>>>(
            buf0, g_b[l], (l < 2) ? bufbf : (unsigned short*)nullptr, N, HC, relu[l]);
    }

    // ---- decoder: out[N,3] = buf0[N,128] @ dec_W + dec_b ----
    decoder_kernel<<<nblk((long long)N * 3, B), B, 0, stream>>>(buf0, dec_W, dec_b, out, N);
}
